// DA_attention_68521908240978
// MI455X (gfx1250) — matrix operand order repaired and checked by reading, not compiled, not run on hardware
//
#include <hip/hip_runtime.h>
#include <hip/hip_bf16.h>

typedef __attribute__((ext_vector_type(16))) _Float16 v16h;
typedef __attribute__((ext_vector_type(8)))  float    v8f;
typedef __attribute__((ext_vector_type(4)))  unsigned int u32x4;
typedef __attribute__((ext_vector_type(8)))  int          i32x8;
typedef __attribute__((ext_vector_type(4)))  int          i32x4;

#define N_TOK 4096
#define CDIM  96
#define HDIM  384

#if defined(__has_builtin)
#if __has_builtin(__builtin_amdgcn_tensor_load_to_lds)
#define HAVE_TDM 1
#endif
#if __has_builtin(__builtin_amdgcn_s_wait_tensorcnt)
#define HAVE_TCNT 1
#endif
#endif

// ---- WMMA lane/VGPR layout helpers (CDNA5 ISA, 16-bit A 16x32) ----
__device__ __forceinline__ int a_kmap(int e, int half) {
  return half * 16 + e;
}
// B-matrix 32x16 (k order used here): lanes 0-15 rows K=0..15 (row=e), lanes 16-31 rows 16..31; col N=lane&15.
// C/D 16x16 f32: VGPR r -> M = r + 8*(lane>>4), N = lane&15.

__device__ __forceinline__ float gelu_exact(float x) {
  return 0.5f * x * (1.0f + erff(x * 0.70710678118654752f));
}

// ------------------------------------------------------------------
// TDM: stage a [rows x rowlen] f16 tile (row stride in elements) into LDS.
// D# per CDNA5 ISA: group0 {count=1, lds_addr,
// global_addr, type=2}, group1 {data_size=2B, tensor_dim0/1, tile_dim0/1,
// tensor_dim0_stride}; 2D tensor -> groups 2/3 zero. Tracked by TENSORcnt.
// 6-arg builtin variant (therock/clang-23 headers).
// ------------------------------------------------------------------
__device__ __forceinline__ void tdm_load_tile(const _Float16* gsrc,
                                              unsigned lds_byte, int rows,
                                              int rowlen, int row_stride) {
#ifdef HAVE_TDM
  unsigned long long ga = (unsigned long long)(size_t)gsrc;
  u32x4 g0;
  g0[0] = 1u;                                            // count=1 (valid)
  g0[1] = lds_byte;                                      // lds_addr (bytes)
  g0[2] = (unsigned)(ga & 0xffffffffu);                  // global_addr lo
  g0[3] = (unsigned)((ga >> 32) & 0x01ffffffu) | (2u << 30);  // hi | type=2
  unsigned td0 = (unsigned)row_stride;                   // tensor_dim0
  unsigned td1 = (unsigned)N_TOK;                        // tensor_dim1
  unsigned long long s0 = (unsigned long long)row_stride;
  i32x8 g1;
  g1[0] = (int)(1u << 16);                               // data_size=2 bytes
  g1[1] = (int)((td0 & 0xffffu) << 16);                  // td0[15:0] -> bits63:48
  g1[2] = (int)((td0 >> 16) | ((td1 & 0xffffu) << 16));
  g1[3] = (int)((td1 >> 16) | ((unsigned)rowlen << 16)); // tile_dim0
  g1[4] = (int)(unsigned)rows;                           // tile_dim1, tile_dim2=0
  g1[5] = (int)(unsigned)(s0 & 0xffffffffu);             // dim0 stride lo
  g1[6] = (int)(unsigned)((s0 >> 32) & 0xffffu);         // stride hi, dim1 stride=0
  g1[7] = 0;
  i32x4 z4 = {0, 0, 0, 0};
  i32x8 z8 = {0, 0, 0, 0, 0, 0, 0, 0};
  __builtin_amdgcn_tensor_load_to_lds(g0, g1, z4, z4, z8, 0);
#else
  (void)gsrc; (void)lds_byte; (void)rows; (void)rowlen; (void)row_stride;
#endif
}

__device__ __forceinline__ void wait_tensorcnt0() {
#ifdef HAVE_TCNT
  __builtin_amdgcn_s_wait_tensorcnt(0);
#endif
}

// ------------------------------------------------------------------
// layout shuffles
// ------------------------------------------------------------------
__global__ __launch_bounds__(256) void k_chw_to_nc(const float* __restrict__ x,
                                                   float* __restrict__ t) {
  int idx = blockIdx.x * 256 + threadIdx.x;
  if (idx >= N_TOK * CDIM) return;
  int n = idx / CDIM, c = idx % CDIM;
  t[idx] = x[(size_t)c * N_TOK + n];
}

__global__ __launch_bounds__(256) void k_nc_to_chw(const float* __restrict__ t,
                                                   float* __restrict__ o) {
  int idx = blockIdx.x * 256 + threadIdx.x;
  if (idx >= N_TOK * CDIM) return;
  int n = idx / CDIM, c = idx % CDIM;
  o[(size_t)c * N_TOK + n] = t[idx];
}

// ------------------------------------------------------------------
// Weight pre-pack: w [O,K] f32 -> WMMA B-fragment order, f16.
// Tile (kt=k0/32, nt=n0/16) at tile*512; lane L, elem e holds
// B[k0+(L>>4)*16+e][n0+(L&15)] so GEMM loads B as two b128s per lane.
// ------------------------------------------------------------------
__global__ __launch_bounds__(256) void k_wpack(const float* __restrict__ w,
                                               _Float16* __restrict__ wp,
                                               int O, int K) {
  int idx = blockIdx.x * 256 + threadIdx.x;
  int tot = (K >> 5) * (O >> 4) * 512;
  if (idx >= tot) return;
  int tile = idx >> 9, within = idx & 511;
  int lane = within >> 4, e = within & 15;
  int ntiles = O >> 4;
  int kt = tile / ntiles, nt = tile % ntiles;
  int row = kt * 32 + (lane >> 4) * 16 + e;
  int coln = nt * 16 + (lane & 15);
  wp[idx] = (_Float16)w[(size_t)coln * K + row];
}

// f16 [N, stride] (column offset coff, width CDIM) -> f16 [CDIM, N]
__global__ __launch_bounds__(256) void k_vtrans(const _Float16* __restrict__ src,
                                                int stride, int coff,
                                                _Float16* __restrict__ dst) {
  int idx = blockIdx.x * 256 + threadIdx.x;
  if (idx >= CDIM * N_TOK) return;
  int c = idx / N_TOK, n = idx % N_TOK;
  dst[idx] = src[(size_t)n * stride + coff + c];
}

// ------------------------------------------------------------------
// LayerNorm over C=96, one wave per row (wave32: 3 elems/lane)
// ------------------------------------------------------------------
__global__ __launch_bounds__(256) void k_layernorm(const float* __restrict__ X,
                                                   const float* __restrict__ w,
                                                   const float* __restrict__ b,
                                                   float* __restrict__ Yf,
                                                   _Float16* __restrict__ Yh) {
  int row = blockIdx.x * 8 + (threadIdx.x >> 5);
  int lane = threadIdx.x & 31;
  if (row >= N_TOK) return;
  const float* x = X + (size_t)row * CDIM;
  float v0 = x[lane], v1 = x[lane + 32], v2 = x[lane + 64];
  float s = v0 + v1 + v2;
#pragma unroll
  for (int off = 1; off < 32; off <<= 1) s += __shfl_xor(s, off, 32);
  float mean = s * (1.0f / CDIM);
  float d0 = v0 - mean, d1 = v1 - mean, d2 = v2 - mean;
  float q = d0 * d0 + d1 * d1 + d2 * d2;
#pragma unroll
  for (int off = 1; off < 32; off <<= 1) q += __shfl_xor(q, off, 32);
  float inv = rsqrtf(q * (1.0f / CDIM) + 1e-5f);
  float y0 = d0 * inv * w[lane] + b[lane];
  float y1 = d1 * inv * w[lane + 32] + b[lane + 32];
  float y2 = d2 * inv * w[lane + 64] + b[lane + 64];
  float* yf = Yf + (size_t)row * CDIM;
  _Float16* yh = Yh + (size_t)row * CDIM;
  yf[lane] = y0; yf[lane + 32] = y1; yf[lane + 64] = y2;
  yh[lane] = (_Float16)y0; yh[lane + 32] = (_Float16)y1; yh[lane + 64] = (_Float16)y2;
}

// ------------------------------------------------------------------
// WMMA GEMM, 32x32 output per wave (2x2 tiles, 4 WMMAs / k-step).
// ------------------------------------------------------------------
__global__ __launch_bounds__(32) void k_gemm_wmma(const _Float16* __restrict__ A,
                                                  const _Float16* __restrict__ Bp,
                                                  const float* __restrict__ bias,
                                                  const float* __restrict__ resid,
                                                  float* __restrict__ Cf,
                                                  _Float16* __restrict__ Ch,
                                                  int M, int K, int Nn, int act) {
  int lane = threadIdx.x;
  int half = lane >> 4, col = lane & 15;
  int m0 = blockIdx.y * 32, n0 = blockIdx.x * 32;
  int ntiles = Nn >> 4;
  v8f acc00 = {}, acc01 = {}, acc10 = {}, acc11 = {};
  const _Float16* arow0 = A + (size_t)(m0 + col) * K;
  const _Float16* arow1 = A + (size_t)(m0 + 16 + col) * K;
  for (int k0 = 0; k0 < K; k0 += 32) {
    __builtin_prefetch(arow0 + k0 + 64, 0, 3);
    __builtin_prefetch(arow1 + k0 + 64, 0, 3);
    v16h aLo, aHi;
#pragma unroll
    for (int e = 0; e < 16; ++e) {
      int kk = k0 + a_kmap(e, half);
      aLo[e] = arow0[kk];
      aHi[e] = arow1[kk];
    }
    const _Float16* bbase =
        Bp + (((size_t)(k0 >> 5)) * ntiles + (n0 >> 4)) * 512 + lane * 16;
    v16h b0 = *(const v16h*)bbase;
    v16h b1 = *(const v16h*)(bbase + 512);
    acc00 = __builtin_amdgcn_wmma_f32_16x16x32_f16(false, aLo, false, b0, (short)0,
                                                   acc00, false, false);
    acc01 = __builtin_amdgcn_wmma_f32_16x16x32_f16(false, aLo, false, b1, (short)0,
                                                   acc01, false, false);
    acc10 = __builtin_amdgcn_wmma_f32_16x16x32_f16(false, aHi, false, b0, (short)0,
                                                   acc10, false, false);
    acc11 = __builtin_amdgcn_wmma_f32_16x16x32_f16(false, aHi, false, b1, (short)0,
                                                   acc11, false, false);
  }
  v8f accs[2][2] = {{acc00, acc01}, {acc10, acc11}};
#pragma unroll
  for (int ti = 0; ti < 2; ++ti) {
#pragma unroll
    for (int r = 0; r < 8; ++r) {
      int m = m0 + ti * 16 + r + 8 * half;
#pragma unroll
      for (int tj = 0; tj < 2; ++tj) {
        int n = n0 + tj * 16 + col;
        float v = accs[ti][tj][r];
        if (bias) v += bias[n];
        if (act == 1) v = gelu_exact(v);
        if (resid) v += resid[(size_t)m * Nn + n];
        if (Cf) Cf[(size_t)m * Nn + n] = v;
        if (Ch) Ch[(size_t)m * Nn + n] = (_Float16)v;
      }
    }
  }
}

// ------------------------------------------------------------------
// Flash-style WMMA attention with TDM-staged K tiles (double-buffered).
// Q/K: [N_TOK, stride] f16, head h at column h*dh (dh <= 32).
// Vt: [CDIM, N_TOK] f16 so PV B-fragments load contiguously.
// One wave per (16-query block, head); 16-key tiles, online softmax;
// P tile round-trips LDS for D->A layout conversion.
// ------------------------------------------------------------------
__global__ __launch_bounds__(32) void k_flash_wmma(const _Float16* __restrict__ Q,
                                                   const _Float16* __restrict__ Km,
                                                   const _Float16* __restrict__ Vt,
                                                   _Float16* __restrict__ O,
                                                   int stride, int dh, float scale) {
  int qb = blockIdx.x, head = blockIdx.y;
  int hoff = head * dh;
  int lane = threadIdx.x, half = lane >> 4, col = lane & 15;
  __shared__ _Float16 pls[16 * 16];
#ifdef HAVE_TDM
  __shared__ _Float16 kstage[2][16 * 32];      // [buf][key*dh + d], dh<=32
  unsigned klsb[2];
  klsb[0] = (unsigned)(size_t)(void*)&kstage[0][0];
  klsb[1] = (unsigned)(size_t)(void*)&kstage[1][0];
#endif

  v16h qa;
  const _Float16* qrow = Q + (size_t)(qb * 16 + col) * stride + hoff;
#pragma unroll
  for (int e = 0; e < 16; ++e) {
    int kk = a_kmap(e, half);
    qa[e] = (kk < dh) ? qrow[kk] : (_Float16)0.0f;
  }

  float mrow[8], lrow[8];
#pragma unroll
  for (int r = 0; r < 8; ++r) { mrow[r] = -1e30f; lrow[r] = 0.0f; }
  v8f acc0 = {}, acc1 = {};

  int d1row = hoff + 16 + col;
  if (d1row > CDIM - 1) d1row = CDIM - 1;
  bool d1ok = (16 + col) < dh;

#ifdef HAVE_TDM
  tdm_load_tile(Km + hoff, klsb[0], 16, dh, stride);   // prime buffer 0
  int buf = 0;
#endif

  for (int kt = 0; kt < N_TOK; kt += 16) {
    v16h kb;
#ifdef HAVE_TDM
    wait_tensorcnt0();                       // current buffer resident
    __syncthreads();
    if (kt + 16 < N_TOK)                     // async-prefetch next tile
      tdm_load_tile(Km + (size_t)(kt + 16) * stride + hoff,
                    klsb[buf ^ 1], 16, dh, stride);
    const _Float16* klds = &kstage[buf][0];
#pragma unroll
    for (int e = 0; e < 16; ++e) {
      int d = half * 16 + e;
      kb[e] = (d < dh) ? klds[col * dh + d] : (_Float16)0.0f;
    }
    buf ^= 1;
#else
    const _Float16* krow = Km + (size_t)(kt + col) * stride + hoff + half * 16;
#pragma unroll
    for (int e = 0; e < 16; ++e) {
      int d = half * 16 + e;
      kb[e] = (d < dh) ? krow[e] : (_Float16)0.0f;
    }
#endif
    v8f s = {};
    s = __builtin_amdgcn_wmma_f32_16x16x32_f16(false, qa, false, kb, (short)0,
                                               s, false, false);
    __syncthreads();
    float alpha[8];
#pragma unroll
    for (int r = 0; r < 8; ++r) {
      float sv = s[r] * scale;
      float mx = sv;
#pragma unroll
      for (int off = 1; off < 16; off <<= 1) mx = fmaxf(mx, __shfl_xor(mx, off, 32));
      float mnew = fmaxf(mrow[r], mx);
      float pv = __expf(sv - mnew);
      float ps = pv;
#pragma unroll
      for (int off = 1; off < 16; off <<= 1) ps += __shfl_xor(ps, off, 32);
      float al = __expf(mrow[r] - mnew);
      lrow[r] = lrow[r] * al + ps;
      mrow[r] = mnew;
      alpha[r] = al;
      pls[(r + 8 * half) * 16 + col] = (_Float16)pv;   // [query M][key]
    }
    __syncthreads();
    v16h pa;
#pragma unroll
    for (int e = 0; e < 16; ++e) {
      int kk = a_kmap(e, half);
      pa[e] = (kk < 16) ? pls[col * 16 + kk] : (_Float16)0.0f;
    }
#pragma unroll
    for (int r = 0; r < 8; ++r) { acc0[r] *= alpha[r]; acc1[r] *= alpha[r]; }
    {
      v16h vb;
      const _Float16* vrow = Vt + (size_t)(hoff + col) * N_TOK + kt;
#pragma unroll
      for (int e = 0; e < 16; ++e)
        vb[e] = (half == 0) ? vrow[e] : (_Float16)0.0f;
      acc0 = __builtin_amdgcn_wmma_f32_16x16x32_f16(false, pa, false, vb, (short)0,
                                                    acc0, false, false);
    }
    if (dh > 16) {
      v16h vb;
      const _Float16* vrow = Vt + (size_t)d1row * N_TOK + kt;
#pragma unroll
      for (int e = 0; e < 16; ++e)
        vb[e] = (half == 0 && d1ok) ? vrow[e] : (_Float16)0.0f;
      acc1 = __builtin_amdgcn_wmma_f32_16x16x32_f16(false, pa, false, vb, (short)0,
                                                    acc1, false, false);
    }
  }
#pragma unroll
  for (int r = 0; r < 8; ++r) {
    int m = qb * 16 + r + 8 * half;
    float inv = 1.0f / lrow[r];
    O[(size_t)m * CDIM + hoff + col] = (_Float16)(acc0[r] * inv);
    if (dh > 16 && d1ok)
      O[(size_t)m * CDIM + hoff + 16 + col] = (_Float16)(acc1[r] * inv);
  }
}

// ------------------------------------------------------------------
// Deformable offset net: depthwise 7x7 -> LN(48) -> GELU -> 1x1(->2) -> tanh
// ------------------------------------------------------------------
__global__ __launch_bounds__(256) void k_offset(const float* __restrict__ q,
                                                const float* __restrict__ dww,
                                                const float* __restrict__ dwb,
                                                const float* __restrict__ lnw,
                                                const float* __restrict__ lnb,
                                                const float* __restrict__ pw,
                                                float* __restrict__ pos) {
  int idx = blockIdx.x * 256 + threadIdx.x;
  if (idx >= 2 * N_TOK) return;
  int g = idx >> 12, p = idx & 4095;
  int y = p >> 6, x = p & 63;
  float acc[48];
  for (int c = 0; c < 48; ++c) {
    float s = 0.0f;
    for (int dy = 0; dy < 7; ++dy) {
      int yy = y + dy - 3;
      if (yy < 0 || yy >= 64) continue;
      for (int dx = 0; dx < 7; ++dx) {
        int xx = x + dx - 3;
        if (xx < 0 || xx >= 64) continue;
        s += dww[c * 49 + dy * 7 + dx] * q[(size_t)(yy * 64 + xx) * CDIM + g * 48 + c];
      }
    }
    acc[c] = s + dwb[c];
  }
  float mean = 0.0f;
  for (int c = 0; c < 48; ++c) mean += acc[c];
  mean *= (1.0f / 48.0f);
  float var = 0.0f;
  for (int c = 0; c < 48; ++c) { float d = acc[c] - mean; var += d * d; }
  var *= (1.0f / 48.0f);
  float inv = rsqrtf(var + 1e-5f);
  float o0 = 0.0f, o1 = 0.0f;
  for (int c = 0; c < 48; ++c) {
    float h = (acc[c] - mean) * inv * lnw[c] + lnb[c];
    h = gelu_exact(h);
    o0 += pw[c] * h;
    o1 += pw[48 + c] * h;
  }
  float offy = tanhf(o0) * (2.0f / 64.0f);
  float offx = tanhf(o1) * (2.0f / 64.0f);
  float refy = ((y + 0.5f) / 64.0f) * 2.0f - 1.0f;
  float refx = ((x + 0.5f) / 64.0f) * 2.0f - 1.0f;
  pos[(size_t)idx * 2 + 0] = offy + refy;
  pos[(size_t)idx * 2 + 1] = offx + refx;
}

// ------------------------------------------------------------------
// Bilinear grid-sample (align_corners, zero-pad) from img=[N_TOK,96] f32
// ------------------------------------------------------------------
__global__ __launch_bounds__(256) void k_gridsample(const float* __restrict__ img,
                                                    const float* __restrict__ pos,
                                                    _Float16* __restrict__ xs) {
  int idx = blockIdx.x * 256 + threadIdx.x;
  if (idx >= 2 * N_TOK) return;
  int g = idx >> 12, n = idx & 4095;
  float fy = (pos[(size_t)idx * 2 + 0] + 1.0f) * 0.5f * 63.0f;
  float fx = (pos[(size_t)idx * 2 + 1] + 1.0f) * 0.5f * 63.0f;
  float y0f = floorf(fy), x0f = floorf(fx);
  int y0 = (int)y0f, x0 = (int)x0f;
  float wy1 = fy - y0f, wy0 = 1.0f - wy1;
  float wx1 = fx - x0f, wx0 = 1.0f - wx1;
  int yi[2] = {y0, y0 + 1}, xi[2] = {x0, x0 + 1};
  float wy[2] = {wy0, wy1}, wx[2] = {wx0, wx1};
  int base[4]; float wgt[4];
#pragma unroll
  for (int a = 0; a < 2; ++a)
#pragma unroll
    for (int b = 0; b < 2; ++b) {
      int yy = yi[a], xx = xi[b];
      float valid = (yy >= 0 && yy < 64 && xx >= 0 && xx < 64) ? 1.0f : 0.0f;
      int yc = min(max(yy, 0), 63), xc = min(max(xx, 0), 63);
      base[a * 2 + b] = (yc * 64 + xc) * CDIM + g * 48;
      wgt[a * 2 + b] = wy[a] * wx[b] * valid;
    }
  for (int c = 0; c < 48; ++c) {
    float v = wgt[0] * img[base[0] + c] + wgt[1] * img[base[1] + c] +
              wgt[2] * img[base[2] + c] + wgt[3] * img[base[3] + c];
    xs[(size_t)n * CDIM + g * 48 + c] = (_Float16)v;
  }
}

// ------------------------------------------------------------------
extern "C" void kernel_launch(void* const* d_in, const int* in_sizes, int n_in,
                              void* d_out, int out_size, void* d_ws, size_t ws_size,
                              hipStream_t stream) {
  (void)in_sizes; (void)n_in; (void)out_size; (void)ws_size;
  const float* x       = (const float*)d_in[0];
  const float* ln1_w   = (const float*)d_in[1];
  const float* ln1_b   = (const float*)d_in[2];
  const float* qkv_w   = (const float*)d_in[3];
  const float* out_w   = (const float*)d_in[4];
  const float* out_b   = (const float*)d_in[5];
  const float* ln2_w   = (const float*)d_in[6];
  const float* ln2_b   = (const float*)d_in[7];
  const float* fc1a_w  = (const float*)d_in[8];
  const float* fc1a_b  = (const float*)d_in[9];
  const float* fc2a_w  = (const float*)d_in[10];
  const float* fc2a_b  = (const float*)d_in[11];
  const float* ln3_w   = (const float*)d_in[12];
  const float* ln3_b   = (const float*)d_in[13];
  const float* dq_w    = (const float*)d_in[14];
  const float* dq_b    = (const float*)d_in[15];
  const float* dk_w    = (const float*)d_in[16];
  const float* dk_b    = (const float*)d_in[17];
  const float* dv_w    = (const float*)d_in[18];
  const float* dv_b    = (const float*)d_in[19];
  const float* do_w    = (const float*)d_in[20];
  const float* do_b    = (const float*)d_in[21];
  const float* off_dw_w= (const float*)d_in[22];
  const float* off_dw_b= (const float*)d_in[23];
  const float* off_ln_w= (const float*)d_in[24];
  const float* off_ln_b= (const float*)d_in[25];
  const float* off_pw_w= (const float*)d_in[26];
  const float* ln4_w   = (const float*)d_in[27];
  const float* ln4_b   = (const float*)d_in[28];
  const float* fc1b_w  = (const float*)d_in[29];
  const float* fc1b_b  = (const float*)d_in[30];
  const float* fc2b_w  = (const float*)d_in[31];
  const float* fc2b_b  = (const float*)d_in[32];

  char* p = (char*)d_ws;
  auto alloc = [&](size_t b) { char* r = p; p += (b + 255) & ~(size_t)255; return r; };
  const size_t NC4 = (size_t)N_TOK * CDIM * 4, NC2 = (size_t)N_TOK * CDIM * 2;
  float*    tA    = (float*)alloc(NC4);
  float*    tB    = (float*)alloc(NC4);
  float*    h32   = (float*)alloc(NC4);
  _Float16* h16   = (_Float16*)alloc(NC2);
  float*    q32   = (float*)alloc(NC4);
  _Float16* q16   = (_Float16*)alloc(NC2);
  _Float16* o16   = (_Float16*)alloc(NC2);
  _Float16* xs16  = (_Float16*)alloc(NC2);
  _Float16* k16   = (_Float16*)alloc(NC2);
  _Float16* v16_  = (_Float16*)alloc(NC2);
  _Float16* vt16  = (_Float16*)alloc(NC2);
  _Float16* qkv16 = (_Float16*)alloc((size_t)N_TOK * 3 * CDIM * 2);
  _Float16* m16   = (_Float16*)alloc((size_t)N_TOK * HDIM * 2);
  float*    big32 = (float*)alloc((size_t)N_TOK * HDIM * 4);
  float*    pos   = (float*)alloc((size_t)2 * N_TOK * 2 * 4);
  _Float16* qkvP  = (_Float16*)alloc((size_t)CDIM * 3 * CDIM * 2);
  _Float16* outP  = (_Float16*)alloc((size_t)CDIM * CDIM * 2);
  _Float16* fc1aP = (_Float16*)alloc((size_t)CDIM * HDIM * 2);
  _Float16* fc2aP = (_Float16*)alloc((size_t)HDIM * CDIM * 2);
  _Float16* dqP   = (_Float16*)alloc((size_t)CDIM * CDIM * 2);
  _Float16* dkP   = (_Float16*)alloc((size_t)CDIM * CDIM * 2);
  _Float16* dvP   = (_Float16*)alloc((size_t)CDIM * CDIM * 2);
  _Float16* doP   = (_Float16*)alloc((size_t)CDIM * CDIM * 2);
  _Float16* fc1bP = (_Float16*)alloc((size_t)CDIM * HDIM * 2);
  _Float16* fc2bP = (_Float16*)alloc((size_t)HDIM * CDIM * 2);

  auto wp = [&](const float* src, _Float16* dst, int O, int K) {
    int tot = (K >> 5) * (O >> 4) * 512;
    k_wpack<<<dim3((tot + 255) / 256), dim3(256), 0, stream>>>(src, dst, O, K);
  };
  auto gemm = [&](const _Float16* A, const _Float16* B, const float* bias,
                  const float* resid, float* Cf, _Float16* Ch,
                  int M, int K, int Nn, int act) {
    k_gemm_wmma<<<dim3(Nn / 32, M / 32), dim3(32), 0, stream>>>(
        A, B, bias, resid, Cf, Ch, M, K, Nn, act);
  };
  auto ln = [&](const float* X, const float* w, const float* b,
                float* Yf, _Float16* Yh) {
    k_layernorm<<<dim3(N_TOK / 8), dim3(256), 0, stream>>>(X, w, b, Yf, Yh);
  };
  auto vtr = [&](const _Float16* src, int stride, int coff, _Float16* dst) {
    k_vtrans<<<dim3((CDIM * N_TOK + 255) / 256), dim3(256), 0, stream>>>(
        src, stride, coff, dst);
  };

  wp(qkv_w, qkvP, 3 * CDIM, CDIM);
  wp(out_w, outP, CDIM, CDIM);
  wp(fc1a_w, fc1aP, HDIM, CDIM);
  wp(fc2a_w, fc2aP, CDIM, HDIM);
  wp(dq_w, dqP, CDIM, CDIM);
  wp(dk_w, dkP, CDIM, CDIM);
  wp(dv_w, dvP, CDIM, CDIM);
  wp(do_w, doP, CDIM, CDIM);
  wp(fc1b_w, fc1bP, HDIM, CDIM);
  wp(fc2b_w, fc2bP, CDIM, HDIM);

  k_chw_to_nc<<<dim3((N_TOK * CDIM + 255) / 256), dim3(256), 0, stream>>>(x, tA);

  // block 1: LN + MHA + residual
  ln(tA, ln1_w, ln1_b, h32, h16);
  gemm(h16, qkvP, nullptr, nullptr, nullptr, qkv16, N_TOK, CDIM, 3 * CDIM, 0);
  vtr(qkv16, 3 * CDIM, 2 * CDIM, vt16);
  k_flash_wmma<<<dim3(N_TOK / 16, 6), dim3(32), 0, stream>>>(
      qkv16 + 0, qkv16 + CDIM, vt16, o16, 3 * CDIM, 16, 0.25f);
  gemm(o16, outP, out_b, tA, tB, nullptr, N_TOK, CDIM, CDIM, 0);

  // block 2: LN + MLP + residual
  ln(tB, ln2_w, ln2_b, h32, h16);
  gemm(h16, fc1aP, fc1a_b, nullptr, nullptr, m16, N_TOK, CDIM, HDIM, 1);
  gemm(m16, fc2aP, fc2a_b, tB, tA, nullptr, N_TOK, HDIM, CDIM, 0);

  // deformable attention: tA is x0
  ln(tA, ln3_w, ln3_b, h32, h16);
  gemm(h16, dqP, dq_b, nullptr, q32, q16, N_TOK, CDIM, CDIM, 0);
  k_offset<<<dim3((2 * N_TOK + 255) / 256), dim3(256), 0, stream>>>(
      q32, off_dw_w, off_dw_b, off_ln_w, off_ln_b, off_pw_w, pos);
  k_gridsample<<<dim3((2 * N_TOK + 255) / 256), dim3(256), 0, stream>>>(
      h32, pos, xs16);
  gemm(xs16, dkP, dk_b, nullptr, big32, k16, N_TOK, CDIM, CDIM, 0);
  gemm(xs16, dvP, dv_b, nullptr, big32, v16_, N_TOK, CDIM, CDIM, 0);
  vtr(v16_, CDIM, 0, vt16);
  k_flash_wmma<<<dim3(N_TOK / 16, 4), dim3(32), 0, stream>>>(
      q16, k16, vt16, o16, CDIM, 24, 0.20412414523193154f);
  gemm(o16, doP, do_b, tA, tB, nullptr, N_TOK, CDIM, CDIM, 0);

  // block 3: LN + MLP + residual
  ln(tB, ln4_w, ln4_b, h32, h16);
  gemm(h16, fc1bP, fc1b_b, nullptr, nullptr, m16, N_TOK, CDIM, HDIM, 1);
  gemm(m16, fc2bP, fc2b_b, tB, tA, nullptr, N_TOK, HDIM, CDIM, 0);

  k_nc_to_chw<<<dim3((N_TOK * CDIM + 255) / 256), dim3(256), 0, stream>>>(
      tA, (float*)d_out);
}
